// Gemma2DecoderLayerWithExpert_41893111005338
// MI455X (gfx1250) — matrix operand order repaired and checked by reading, not compiled, not run on hardware
//
#include <hip/hip_runtime.h>

typedef __attribute__((ext_vector_type(16))) __bf16        v16bf;
typedef __attribute__((ext_vector_type(8)))  float         v8f;
typedef __attribute__((ext_vector_type(4)))  unsigned int  v4u;

#define kB   2
#define kLP  1024
#define kLE  1024
#define kL   2048
#define kDP  2304
#define kDE  1024
#define kH   8
#define kHKV 4
#define kDH  256
#define kNQ  2048   /* H*DH  */
#define kNKV 1024   /* HKV*DH */
#define SCALE_F 0.0625f

union FragU { v4u u[2]; v16bf v; };

__device__ __forceinline__ void lds_fence() {
#if __has_builtin(__builtin_amdgcn_s_wait_dscnt)
  __builtin_amdgcn_s_wait_dscnt(0);
#endif
  // Same-wave LDS ops are architecturally in-order (CDNA5 ISA §7.3); the
  // compiler also inserts s_wait_dscnt for the aliasing shared-memory load.
}

__device__ __forceinline__ unsigned short f2bf(float f) {
  unsigned int u = __float_as_uint(f);
  u += 0x7FFFu + ((u >> 16) & 1u);
  return (unsigned short)(u >> 16);
}

__device__ __forceinline__ v8f wmma_bf16(v16bf a, v16bf b, v8f c) {
  return __builtin_amdgcn_wmma_f32_16x16x32_bf16(false, a, false, b, (short)0, c,
                                                 false, false);
}

// A-operand fragment 16(M)x32(K) from row-major src (ld in bf16 elements).
// ISA layout: lanes 0-15 -> row, K chunks [0..7],[16..23]; lanes 16-31 -> [8..15],[24..31].
__device__ __forceinline__ v16bf load_a(const unsigned short* base, int ld) {
  int lane = threadIdx.x & 31;
  int r = lane & 15, hf = lane >> 4;
  const unsigned short* p = (const unsigned short*)
      __builtin_assume_aligned(base + (size_t)r * ld + (hf ? 8 : 0), 16);
  FragU f;
  __builtin_memcpy(&f.u[0], p, 16);
  __builtin_memcpy(&f.u[1], p + 16, 16);
  return f.v;
}

// B-operand fragment 32(K)x16(N) from [N][K] row-major src (column n contiguous in K).
// ISA layout: lanes 0-15 -> col n, K chunks [0..7],[16..23]; lanes 16-31 -> [8..15],[24..31].
__device__ __forceinline__ v16bf load_b(const unsigned short* base, int ld) {
  int lane = threadIdx.x & 31;
  int n = lane & 15, hf = lane >> 4;
  const unsigned short* p = (const unsigned short*)
      __builtin_assume_aligned(base + (size_t)n * ld + hf * 8, 16);
  FragU f;
  __builtin_memcpy(&f.u[0], p, 16);
  __builtin_memcpy(&f.u[1], p + 16, 16);
  return f.v;
}

__global__ __launch_bounds__(256) void cvt_bf16(const float* __restrict__ in,
                                                unsigned short* __restrict__ out,
                                                int n) {
  int i = blockIdx.x * 256 + threadIdx.x;
  if (i < n) out[i] = f2bf(in[i]);
}

// in: [K][N] f32  ->  out: [N][K] bf16
__global__ __launch_bounds__(256) void transpose_bf16(const float* __restrict__ in,
                                                      unsigned short* __restrict__ out,
                                                      int K, int N) {
  int i = blockIdx.x * 256 + threadIdx.x;
  if (i >= K * N) return;
  int n = i / K, k = i - n * K;
  out[i] = f2bf(in[(size_t)k * N + n]);
}

// C[M,N] = A[M,K](bf16) @ Bt[N,K]^T(bf16), f32 accumulate.
// Logical row r: b = r/seg, l = r%seg; A row = b*aStride + l + aOff; C row = b*cStride + l + cOff.
__global__ __launch_bounds__(256) void gemm_bf16_wmma(
    const unsigned short* __restrict__ A, int lda,
    const unsigned short* __restrict__ Bt,
    float* __restrict__ C,
    int K, int N,
    int seg, int aStride, int aOff, int cStride, int cOff) {
  int wave = threadIdx.x >> 5;
  int lane = threadIdx.x & 31;
  int row0 = blockIdx.x * 128 + wave * 16;
  int col0 = blockIdx.y * 64;
  int bb = row0 / seg;
  int l0 = row0 - bb * seg;
  const unsigned short* Ap = A + (size_t)(bb * aStride + l0 + aOff) * lda;
  const unsigned short* Bp = Bt + (size_t)col0 * K;
  v8f acc0 = {}, acc1 = {}, acc2 = {}, acc3 = {};
  for (int k0 = 0; k0 < K; k0 += 32) {
    __builtin_prefetch(Ap + k0 + 128, 0, 1);
    __builtin_prefetch(Bp + k0 + 128, 0, 1);
    v16bf a  = load_a(Ap + k0, lda);
    v16bf b0 = load_b(Bp + k0, K);
    v16bf b1 = load_b(Bp + (size_t)16 * K + k0, K);
    v16bf b2 = load_b(Bp + (size_t)32 * K + k0, K);
    v16bf b3 = load_b(Bp + (size_t)48 * K + k0, K);
    acc0 = wmma_bf16(a, b0, acc0);
    acc1 = wmma_bf16(a, b1, acc1);
    acc2 = wmma_bf16(a, b2, acc2);
    acc3 = wmma_bf16(a, b3, acc3);
  }
  int hf = lane >> 4, n = lane & 15;
  float* Cp = C + (size_t)(bb * cStride + l0 + cOff + 8 * hf) * N + col0 + n;
#pragma unroll
  for (int r = 0; r < 8; ++r) {
    float* cr = Cp + (size_t)r * N;
    cr[0] = acc0[r]; cr[16] = acc1[r]; cr[32] = acc2[r]; cr[48] = acc3[r];
  }
}

// RoPE on q: q f32 [B][L][H*DH] -> qb bf16 [B][H][L][DH]
__global__ __launch_bounds__(256) void rope_q(const float* __restrict__ q,
                                              const float* __restrict__ cosv,
                                              const float* __restrict__ sinv,
                                              unsigned short* __restrict__ qb) {
  int i = blockIdx.x * 256 + threadIdx.x;           // B*L*H*128 = 2^22
  int d = i & 127;
  int h = (i >> 7) & 7;
  int l = (i >> 10) & 2047;
  int b = i >> 21;
  const float* qr = q + (size_t)(b * kL + l) * kNQ + h * kDH;
  float c = cosv[(size_t)(b * kL + l) * 128 + d];
  float s = sinv[(size_t)(b * kL + l) * 128 + d];
  float x1 = qr[d], x2 = qr[d + 128];
  unsigned short* qo = qb + (size_t)((b * kH + h) * kL + l) * kDH;
  qo[d]       = f2bf(x1 * c - x2 * s);
  qo[d + 128] = f2bf(x2 * c + x1 * s);
}

// RoPE on k: k f32 [B][L][HKV*DH] -> kb bf16 [B][HKV][L][DH] and present_k f32 [B][L][HKV][DH]
__global__ __launch_bounds__(256) void rope_k(const float* __restrict__ k,
                                              const float* __restrict__ cosv,
                                              const float* __restrict__ sinv,
                                              unsigned short* __restrict__ kb,
                                              float* __restrict__ pk) {
  int i = blockIdx.x * 256 + threadIdx.x;           // B*L*HKV*128 = 2^21
  int d = i & 127;
  int h = (i >> 7) & 3;
  int l = (i >> 9) & 2047;
  int b = i >> 20;
  const float* kr = k + (size_t)(b * kL + l) * kNKV + h * kDH;
  float c = cosv[(size_t)(b * kL + l) * 128 + d];
  float s = sinv[(size_t)(b * kL + l) * 128 + d];
  float x1 = kr[d], x2 = kr[d + 128];
  float o1 = x1 * c - x2 * s;
  float o2 = x2 * c + x1 * s;
  unsigned short* ko = kb + (size_t)((b * kHKV + h) * kL + l) * kDH;
  ko[d] = f2bf(o1); ko[d + 128] = f2bf(o2);
  float* po = pk + (size_t)((b * kL + l) * kHKV + h) * kDH;
  po[d] = o1; po[d + 128] = o2;
}

// v f32 [B][L][HKV*DH] -> present_v (same layout) + vt bf16 [B][HKV][DH][L]
__global__ __launch_bounds__(256) void v_prep(const float* __restrict__ v,
                                              float* __restrict__ pv,
                                              unsigned short* __restrict__ vt) {
  int i = blockIdx.x * 256 + threadIdx.x;           // B*L*HKV*DH = 2^22
  int d = i & 255;
  int h = (i >> 8) & 3;
  int l = (i >> 10) & 2047;
  int b = i >> 21;
  float val = v[i];
  pv[i] = val;
  vt[(size_t)((b * kHKV + h) * kDH + d) * kL + l] = f2bf(val);
}

// Flash attention: one wave per 16-query tile. qb [B][H][L][DH], kb [B][HKV][L][DH],
// vt [B][HKV][DH][L], mask [B][L][L]; out attn bf16 [B][L][H*DH].
__global__ __launch_bounds__(256) void attn_wmma(
    const unsigned short* __restrict__ qb,
    const unsigned short* __restrict__ kb,
    const unsigned short* __restrict__ vt,
    const float* __restrict__ mask,
    unsigned short* __restrict__ attnb) {
  __shared__ unsigned short pbuf[8][16 * 32];
  int wave = threadIdx.x >> 5;
  int lane = threadIdx.x & 31;
  int n = lane & 15, hf = lane >> 4;
  int t = blockIdx.x * 8 + wave;                    // 0..2047
  int qt = t & 127;                                 // L/16
  int h = (t >> 7) & 7;
  int b = t >> 10;
  int kvh = h >> 1;                                 // H/HKV == 2

  const unsigned short* qbase = qb + (size_t)((b * kH + h) * kL + qt * 16) * kDH;
  v16bf qfr[8];
#pragma unroll
  for (int kk = 0; kk < 8; ++kk) qfr[kk] = load_a(qbase + kk * 32, kDH);

  v8f zero = {};
  v8f o[16];
#pragma unroll
  for (int c = 0; c < 16; ++c) o[c] = zero;

  float m_r[8], l_r[8];
#pragma unroll
  for (int r = 0; r < 8; ++r) { m_r[r] = -3.0e38f; l_r[r] = 0.0f; }

  const float* mrow = mask + (size_t)(b * kL + qt * 16 + 8 * hf) * kL;
  const unsigned short* khead = kb + (size_t)((b * kHKV + kvh) * kL) * kDH;
  const unsigned short* vhead = vt + (size_t)((b * kHKV + kvh) * kDH) * kL;
  unsigned short* pb = pbuf[wave];

  for (int kb0 = 0; kb0 < kL; kb0 += 32) {
    v8f s0 = zero, s1 = zero;
    const unsigned short* kt = khead + (size_t)kb0 * kDH;
#pragma unroll
    for (int kk = 0; kk < 8; ++kk) {
      v16bf b0 = load_b(kt + kk * 32, kDH);
      v16bf b1 = load_b(kt + (size_t)16 * kDH + kk * 32, kDH);
      s0 = wmma_bf16(qfr[kk], b0, s0);
      s1 = wmma_bf16(qfr[kk], b1, s1);
    }
    float mn[8], al[8];
#pragma unroll
    for (int r = 0; r < 8; ++r) {
      float mk0 = mrow[(size_t)r * kL + kb0 + n];
      float mk1 = mrow[(size_t)r * kL + kb0 + 16 + n];
      s0[r] = s0[r] * SCALE_F + mk0;
      s1[r] = s1[r] * SCALE_F + mk1;
      float v = fmaxf(s0[r], s1[r]);
      v = fmaxf(v, __shfl_xor(v, 1, 32));
      v = fmaxf(v, __shfl_xor(v, 2, 32));
      v = fmaxf(v, __shfl_xor(v, 4, 32));
      v = fmaxf(v, __shfl_xor(v, 8, 32));
      mn[r] = fmaxf(m_r[r], v);
      al[r] = __expf(m_r[r] - mn[r]);
      m_r[r] = mn[r];
    }
#pragma unroll
    for (int r = 0; r < 8; ++r) {
      float p0 = __expf(s0[r] - mn[r]);
      float p1 = __expf(s1[r] - mn[r]);
      float rs = p0 + p1;
      rs += __shfl_xor(rs, 1, 32);
      rs += __shfl_xor(rs, 2, 32);
      rs += __shfl_xor(rs, 4, 32);
      rs += __shfl_xor(rs, 8, 32);
      l_r[r] = l_r[r] * al[r] + rs;
      pb[(r + 8 * hf) * 32 + n]      = f2bf(p0);
      pb[(r + 8 * hf) * 32 + 16 + n] = f2bf(p1);
    }
#pragma unroll
    for (int c = 0; c < 16; ++c)
#pragma unroll
      for (int r = 0; r < 8; ++r) o[c][r] *= al[r];
    lds_fence();
    v16bf pf = load_a(pb, 32);
    const unsigned short* vt0 = vhead + kb0;
#pragma unroll
    for (int c = 0; c < 16; ++c) {
      v16bf vf = load_b(vt0 + (size_t)(c * 16) * kL, kL);
      o[c] = wmma_bf16(pf, vf, o[c]);
    }
  }
  float inv_l[8];
#pragma unroll
  for (int r = 0; r < 8; ++r) inv_l[r] = 1.0f / l_r[r];
  unsigned short* ob = attnb + (size_t)(b * kL + qt * 16 + 8 * hf) * kNQ + h * kDH + n;
#pragma unroll
  for (int r = 0; r < 8; ++r)
#pragma unroll
    for (int c = 0; c < 16; ++c)
      ob[(size_t)r * kNQ + c * 16] = f2bf(o[c][r] * inv_l[r]);
}

extern "C" void kernel_launch(void* const* d_in, const int* in_sizes, int n_in,
                              void* d_out, int out_size, void* d_ws, size_t ws_size,
                              hipStream_t stream) {
  (void)in_sizes; (void)n_in; (void)out_size; (void)ws_size;
  const float* pali = (const float*)d_in[0];
  const float* expe = (const float*)d_in[1];
  const float* cosv = (const float*)d_in[2];
  const float* sinv = (const float*)d_in[3];
  const float* mask = (const float*)d_in[4];
  const float* Wq0  = (const float*)d_in[5];
  const float* Wk0  = (const float*)d_in[6];
  const float* Wv0  = (const float*)d_in[7];
  const float* Wo0  = (const float*)d_in[8];
  const float* Wq1  = (const float*)d_in[9];
  const float* Wk1  = (const float*)d_in[10];
  const float* Wv1  = (const float*)d_in[11];
  const float* Wo1  = (const float*)d_in[12];

  float* out0 = (float*)d_out;
  float* out1 = out0 + (size_t)kB * kLP * kDP;
  float* pk   = out1 + (size_t)kB * kLE * kDE;
  float* pv   = pk   + (size_t)kB * kL * kNKV;

  char* cur = (char*)d_ws;
  auto alloc = [&](size_t bytes) {
    void* p = (void*)cur;
    cur += (bytes + 255) & ~(size_t)255;
    return p;
  };
  unsigned short* Xp   = (unsigned short*)alloc((size_t)kB * kLP * kDP * 2);
  unsigned short* Xe   = (unsigned short*)alloc((size_t)kB * kLE * kDE * 2);
  unsigned short* Wq0t = (unsigned short*)alloc((size_t)kDP * kNQ  * 2);
  unsigned short* Wk0t = (unsigned short*)alloc((size_t)kDP * kNKV * 2);
  unsigned short* Wv0t = (unsigned short*)alloc((size_t)kDP * kNKV * 2);
  unsigned short* Wo0t = (unsigned short*)alloc((size_t)kNQ * kDP  * 2);
  unsigned short* Wq1t = (unsigned short*)alloc((size_t)kDE * kNQ  * 2);
  unsigned short* Wk1t = (unsigned short*)alloc((size_t)kDE * kNKV * 2);
  unsigned short* Wv1t = (unsigned short*)alloc((size_t)kDE * kNKV * 2);
  unsigned short* Wo1t = (unsigned short*)alloc((size_t)kNQ * kDE  * 2);
  float* qf = (float*)alloc((size_t)kB * kL * kNQ  * 4);
  float* kf = (float*)alloc((size_t)kB * kL * kNKV * 4);
  float* vf = (float*)alloc((size_t)kB * kL * kNKV * 4);
  unsigned short* qbb   = (unsigned short*)alloc((size_t)kB * kL * kNQ  * 2);
  unsigned short* kbb   = (unsigned short*)alloc((size_t)kB * kL * kNKV * 2);
  unsigned short* vtb   = (unsigned short*)alloc((size_t)kB * kL * kNKV * 2);
  unsigned short* attnb = (unsigned short*)alloc((size_t)kB * kL * kNQ  * 2);

  // 1) precision conversion / weight transposition
  cvt_bf16<<<(kB * kLP * kDP) / 256, 256, 0, stream>>>(pali, Xp, kB * kLP * kDP);
  cvt_bf16<<<(kB * kLE * kDE) / 256, 256, 0, stream>>>(expe, Xe, kB * kLE * kDE);
  transpose_bf16<<<(kDP * kNQ ) / 256, 256, 0, stream>>>(Wq0, Wq0t, kDP, kNQ);
  transpose_bf16<<<(kDP * kNKV) / 256, 256, 0, stream>>>(Wk0, Wk0t, kDP, kNKV);
  transpose_bf16<<<(kDP * kNKV) / 256, 256, 0, stream>>>(Wv0, Wv0t, kDP, kNKV);
  transpose_bf16<<<(kNQ * kDP ) / 256, 256, 0, stream>>>(Wo0, Wo0t, kNQ, kDP);
  transpose_bf16<<<(kDE * kNQ ) / 256, 256, 0, stream>>>(Wq1, Wq1t, kDE, kNQ);
  transpose_bf16<<<(kDE * kNKV) / 256, 256, 0, stream>>>(Wk1, Wk1t, kDE, kNKV);
  transpose_bf16<<<(kDE * kNKV) / 256, 256, 0, stream>>>(Wv1, Wv1t, kDE, kNKV);
  transpose_bf16<<<(kNQ * kDE ) / 256, 256, 0, stream>>>(Wo1, Wo1t, kNQ, kDE);

  // 2) QKV projections (M = 2048 rows per segment pair -> grid.x = 16)
  gemm_bf16_wmma<<<dim3(16, kNQ  / 64), 256, 0, stream>>>(Xp, kDP, Wq0t, qf, kDP, kNQ,  kLP, kLP, 0, kL, 0);
  gemm_bf16_wmma<<<dim3(16, kNQ  / 64), 256, 0, stream>>>(Xe, kDE, Wq1t, qf, kDE, kNQ,  kLE, kLE, 0, kL, kLP);
  gemm_bf16_wmma<<<dim3(16, kNKV / 64), 256, 0, stream>>>(Xp, kDP, Wk0t, kf, kDP, kNKV, kLP, kLP, 0, kL, 0);
  gemm_bf16_wmma<<<dim3(16, kNKV / 64), 256, 0, stream>>>(Xe, kDE, Wk1t, kf, kDE, kNKV, kLE, kLE, 0, kL, kLP);
  gemm_bf16_wmma<<<dim3(16, kNKV / 64), 256, 0, stream>>>(Xp, kDP, Wv0t, vf, kDP, kNKV, kLP, kLP, 0, kL, 0);
  gemm_bf16_wmma<<<dim3(16, kNKV / 64), 256, 0, stream>>>(Xe, kDE, Wv1t, vf, kDE, kNKV, kLE, kLE, 0, kL, kLP);

  // 3) RoPE + KV cache outputs + V transpose
  rope_q<<<(kB * kL * kH * 128) / 256, 256, 0, stream>>>(qf, cosv, sinv, qbb);
  rope_k<<<(kB * kL * kHKV * 128) / 256, 256, 0, stream>>>(kf, cosv, sinv, kbb, pk);
  v_prep<<<(kB * kL * kNKV) / 256, 256, 0, stream>>>(vf, pv, vtb);

  // 4) attention: 2048 query tiles, 8 waves/block
  attn_wmma<<<256, 256, 0, stream>>>(qbb, kbb, vtb, mask, attnb);

  // 5) output projections
  gemm_bf16_wmma<<<dim3(16, kDP / 64), 256, 0, stream>>>(attnb, kNQ, Wo0t, out0, kNQ, kDP, kLP, kL, 0,   kLP, 0);
  gemm_bf16_wmma<<<dim3(16, kDE / 64), 256, 0, stream>>>(attnb, kNQ, Wo1t, out1, kNQ, kDE, kLE, kL, kLP, kLE, 0);
}
